// CausalSGU_20633022890419
// MI455X (gfx1250) — matrix operand order repaired and checked by reading, not compiled, not run on hardware
//
#include <hip/hip_runtime.h>
#include <hip/hip_bf16.h>

// Problem constants
#define DIMX   2048
#define NSEQ   2048
#define HEADS  4
#define DHEAD  256
#define GDIM   1024   // gate channels = DIMX/2
#define BATCH  4
#define LN_EPS 1e-5f

typedef __attribute__((ext_vector_type(16))) __bf16 v16bf;
typedef __attribute__((ext_vector_type(8)))  float  v8f;

// f32 -> bf16, round-to-nearest-even
__device__ __forceinline__ unsigned short bf16_rne(float f) {
    unsigned u = __float_as_uint(f);
    u += 0x7fffu + ((u >> 16) & 1u);
    return (unsigned short)(u >> 16);
}
__device__ __forceinline__ unsigned pack_bf16(float a, float b) {
    unsigned ua = __float_as_uint(a);
    unsigned ub = __float_as_uint(b);
    ua += 0x7fffu + ((ua >> 16) & 1u);
    ub += 0x7fffu + ((ub >> 16) & 1u);
    return (ua >> 16) | (ub & 0xffff0000u);
}

// -----------------------------------------------------------------------------
// Kernel 1: LayerNorm of gate half + bf16 convert + transpose into
// G_t[(b*H+h)*256 + d][n]  (bf16, n contiguous) so the GEMM's B-fragment
// K-pairs are contiguous dwords.
// One wave (32 lanes) per sequence row; 16 rows per 512-thread block.
// -----------------------------------------------------------------------------
__global__ __launch_bounds__(512) void ln_transpose_kernel(
    const float* __restrict__ x,
    const float* __restrict__ gamma,
    const float* __restrict__ beta,
    unsigned short* __restrict__ gt)
{
    __shared__ unsigned short tile[16][GDIM];   // 32 KB

    const int w    = threadIdx.x >> 5;
    const int lane = threadIdx.x & 31;
    const size_t row = (size_t)blockIdx.x * 16 + w;       // global (b,n) row
    const float* gx = x + row * DIMX + GDIM;              // gate half

    float vals[32];
    float s = 0.f;
#pragma unroll
    for (int i = 0; i < 32; ++i) { float v = gx[lane + 32 * i]; vals[i] = v; s += v; }
#pragma unroll
    for (int off = 16; off; off >>= 1) s += __shfl_xor(s, off, 32);
    const float mu = s * (1.0f / GDIM);

    float vsum = 0.f;
#pragma unroll
    for (int i = 0; i < 32; ++i) { float d = vals[i] - mu; vsum += d * d; }
#pragma unroll
    for (int off = 16; off; off >>= 1) vsum += __shfl_xor(vsum, off, 32);
    const float rstd = rsqrtf(vsum * (1.0f / GDIM) + LN_EPS);

#pragma unroll
    for (int i = 0; i < 32; ++i) {
        int j = lane + 32 * i;
        float y = (vals[i] - mu) * rstd * gamma[j] + beta[j];
        tile[w][j] = bf16_rne(y);
    }
    __syncthreads();

    // Transposed write: 1024 d-columns, 512 threads -> 2 iterations.
    const int b  = (int)(blockIdx.x >> 7);          // 128 blocks per batch
    const int nb = (int)((blockIdx.x & 127) * 16);  // n base within batch
#pragma unroll
    for (int it = 0; it < 2; ++it) {
        int d  = it * 512 + threadIdx.x;            // 0..1023
        int h  = d >> 8;
        int dl = d & 255;
        unsigned pk[8];
#pragma unroll
        for (int p = 0; p < 8; ++p)
            pk[p] = (unsigned)tile[2 * p][d] | ((unsigned)tile[2 * p + 1][d] << 16);
        size_t off = ((size_t)((b * HEADS + h) * DHEAD + dl)) * NSEQ + nb;
        uint4* dst = (uint4*)(gt + off);            // 32B aligned (nb % 16 == 0)
        dst[0] = make_uint4(pk[0], pk[1], pk[2], pk[3]);
        dst[1] = make_uint4(pk[4], pk[5], pk[6], pk[7]);
    }
}

// -----------------------------------------------------------------------------
// Kernel 2: causal GEMM per (b,h) using v_wmma_f32_16x16x32_bf16, with fused
// bias add and res-multiply epilogue.
//   OUT[m,d] = sum_{n<=m} W[h][m][n] * G_t[bh][d][n]
// Block: 256 threads = 8 waves in 4(m) x 2(d); block tile = 64(m) x 256(d).
// Wave tile: 16(m) x 128(d) = 8 WMMA accumulators.
// Loop control & diagonal guard are forced into SGPRs (readfirstlane) so the
// compiler emits scalar branches and EXEC provably stays all-ones for WMMA.
// -----------------------------------------------------------------------------
__global__ __launch_bounds__(256) void sgu_gemm_kernel(
    const float* __restrict__ x,
    const float* __restrict__ weight,
    const float* __restrict__ bias,
    const unsigned short* __restrict__ gt,
    float* __restrict__ out)
{
    const int w    = threadIdx.x >> 5;
    const int lane = threadIdx.x & 31;
    // Wave-uniform tile coordinates -> SGPRs (scalar loop control, EXEC untouched)
    const int wm   = __builtin_amdgcn_readfirstlane(w >> 1);  // 0..3 (m sub-tile)
    const int wd   = __builtin_amdgcn_readfirstlane(w & 1);   // 0..1 (d sub-tile)
    const int hl   = lane & 15;       // lane within half-wave
    const int hi   = lane >> 4;       // which half-wave

    const int bh = blockIdx.y;        // b*HEADS + h
    const int h  = bh & (HEADS - 1);
    const int b  = bh >> 2;
    const int m0 = blockIdx.x * 64;

    // A: lane holds row mrow of W[h]; K pairs per ISA A-layout.
    const int mrow = m0 + wm * 16 + hl;
    const float* wrow = weight + ((size_t)h * NSEQ + (size_t)mrow) * NSEQ;

    // B: lane holds column d of G_t; K runs along contiguous n.
    const unsigned short* g0 =
        gt + ((size_t)bh * DHEAD + (size_t)wd * 128 + hl) * NSEQ;

    v8f acc[8] = {};

    const int mbase = m0 + wm * 16;       // SGPR: wave-uniform min row
    const int mtop  = mbase + 15;         // SGPR: wave-uniform max row
    const int kmax  = m0 + 64;

    for (int k0 = 0; k0 < kmax; k0 += 32) {
        if (k0 > mtop) break;                       // scalar: fully above diagonal
        const bool needmask = (k0 + 31) > mbase;    // scalar: diagonal tile

        // ---- A fragment: convert f32 weights -> bf16 pairs ----
        const int kbA = k0 + hi * 8;
        const float4* wp = (const float4*)(wrow + kbA);   // 32B aligned
        float4 a01 = wp[0];        // K = kbA+0..3
        float4 a23 = wp[1];        // K = kbA+4..7
        float4 a45 = wp[4];        // K = kbA+16..19
        float4 a67 = wp[5];        // K = kbA+20..23
        if (needmask) {            // causal mask: keep K <= mrow
            a01.x = (kbA +  0 <= mrow) ? a01.x : 0.f;
            a01.y = (kbA +  1 <= mrow) ? a01.y : 0.f;
            a01.z = (kbA +  2 <= mrow) ? a01.z : 0.f;
            a01.w = (kbA +  3 <= mrow) ? a01.w : 0.f;
            a23.x = (kbA +  4 <= mrow) ? a23.x : 0.f;
            a23.y = (kbA +  5 <= mrow) ? a23.y : 0.f;
            a23.z = (kbA +  6 <= mrow) ? a23.z : 0.f;
            a23.w = (kbA +  7 <= mrow) ? a23.w : 0.f;
            a45.x = (kbA + 16 <= mrow) ? a45.x : 0.f;
            a45.y = (kbA + 17 <= mrow) ? a45.y : 0.f;
            a45.z = (kbA + 18 <= mrow) ? a45.z : 0.f;
            a45.w = (kbA + 19 <= mrow) ? a45.w : 0.f;
            a67.x = (kbA + 20 <= mrow) ? a67.x : 0.f;
            a67.y = (kbA + 21 <= mrow) ? a67.y : 0.f;
            a67.z = (kbA + 22 <= mrow) ? a67.z : 0.f;
            a67.w = (kbA + 23 <= mrow) ? a67.w : 0.f;
        }
        union { unsigned u[8]; v16bf v; } A;
        A.u[0] = pack_bf16(a01.x, a01.y);
        A.u[1] = pack_bf16(a01.z, a01.w);
        A.u[2] = pack_bf16(a23.x, a23.y);
        A.u[3] = pack_bf16(a23.z, a23.w);
        A.u[4] = pack_bf16(a45.x, a45.y);
        A.u[5] = pack_bf16(a45.z, a45.w);
        A.u[6] = pack_bf16(a67.x, a67.y);
        A.u[7] = pack_bf16(a67.z, a67.w);

        // ---- B fragments + WMMA over 8 d-tiles ----
        const int kbB = k0 + hi * 8;
#pragma unroll
        for (int t = 0; t < 8; ++t) {
            const uint4* gp =
                (const uint4*)(g0 + (size_t)t * 16 * NSEQ + kbB);  // 16B aligned
            union { uint4 q[2]; v16bf v; } B;
            B.q[0] = gp[0];
            B.q[1] = gp[2];
            acc[t] = __builtin_amdgcn_wmma_f32_16x16x32_bf16(
                /*neg_a=*/false, A.v, /*neg_b=*/false, B.v,
                /*c_mod=*/(short)0, acc[t],
                /*reuse_a=*/false, /*reuse_b=*/false);
        }
    }

    // ---- Epilogue: + bias[h,m], * res, write f32 out[b, m, h*256+d] ----
#pragma unroll
    for (int t = 0; t < 8; ++t) {
        const int d = wd * 128 + t * 16 + hl;
        const int c = h * DHEAD + d;                 // output channel (0..1023)
#pragma unroll
        for (int r = 0; r < 8; ++r) {
            const int m = m0 + wm * 16 + r + hi * 8; // C layout: vgpr r -> row r / r+8
            float v  = acc[t][r] + bias[h * NSEQ + m];
            float rv = x[((size_t)b * NSEQ + m) * DIMX + c];   // res half
            out[((size_t)b * NSEQ + m) * GDIM + c] = v * rv;
        }
    }
}

extern "C" void kernel_launch(void* const* d_in, const int* in_sizes, int n_in,
                              void* d_out, int out_size, void* d_ws, size_t ws_size,
                              hipStream_t stream) {
    const float* x      = (const float*)d_in[0];
    const float* weight = (const float*)d_in[1];
    const float* bias   = (const float*)d_in[2];
    const float* gamma  = (const float*)d_in[3];
    const float* beta   = (const float*)d_in[4];
    float* out = (float*)d_out;

    // Workspace: G_t bf16 [B*H*256][2048] = 16 MB
    unsigned short* gt = (unsigned short*)d_ws;

    // 1) LayerNorm + transpose: 8192 rows, 16 rows per block of 512 threads.
    ln_transpose_kernel<<<dim3(BATCH * NSEQ / 16), dim3(512), 0, stream>>>(
        x, gamma, beta, gt);

    // 2) Causal WMMA GEMM + fused epilogue: grid (M/64, B*H).
    sgu_gemm_kernel<<<dim3(NSEQ / 64, BATCH * HEADS), dim3(256), 0, stream>>>(
        x, weight, bias, gt, out);

    (void)in_sizes; (void)n_in; (void)out_size; (void)ws_size;
}
